// HAN_3762391352090
// MI455X (gfx1250) — matrix operand order repaired and checked by reading, not compiled, not run on hardware
//
#include <hip/hip_runtime.h>
#include <math.h>

#define NN      100000   // nodes
#define EE      600000   // edges per edge type
#define CD      128      // hidden dim
#define HH      2        // heads
#define DH      64       // per-head dim
#define INF_    64       // input feature dim
#define OUTD    8        // classifier out
#define SLOPE   0.2f

typedef __attribute__((ext_vector_type(16))) _Float16 v16h;
typedef __attribute__((ext_vector_type(8)))  _Float16 v8h;
typedef __attribute__((ext_vector_type(8)))  float    v8f;

// ---------------------------------------------------------------- helpers
__device__ __forceinline__ void atomicMaxF(float* addr, float val) {
    // float max via sign-aware integer atomics (valid incl. -inf init, negatives)
    if (val >= 0.0f) atomicMax((int*)addr, __float_as_int(val));
    else             atomicMin((unsigned int*)addr, (unsigned int)__float_as_int(val));
}

// ---------------------------------------------------------------- converts
__global__ void k_f32_to_f16(const float* __restrict__ x, _Float16* __restrict__ y, long n) {
    long i = (long)blockIdx.x * blockDim.x + threadIdx.x;
    if (i < n) y[i] = (_Float16)x[i];
}

// W: [K][Cout] row-major f32  ->  Bt: [Cout][K] row-major f16
__global__ void k_wtrans_f16(const float* __restrict__ W, _Float16* __restrict__ Bt,
                             int K, int Cout) {
    int idx = blockIdx.x * blockDim.x + threadIdx.x;
    if (idx < K * Cout) {
        int k = idx / Cout, n = idx % Cout;
        Bt[(size_t)n * K + k] = (_Float16)W[idx];
    }
}

__global__ void k_fill(float* __restrict__ p, float v, int n) {
    int i = blockIdx.x * blockDim.x + threadIdx.x;
    if (i < n) p[i] = v;
}

__global__ void k_relu(float* __restrict__ p, long n) {
    long i = (long)blockIdx.x * blockDim.x + threadIdx.x;
    if (i < n) p[i] = fmaxf(p[i], 0.0f);
}

// ---------------------------------------------------------------- WMMA GEMM
// A: [N][K] f16, Bt: [Cout=128][K] f16, bias: [128] f32.
// mode 0: out[N][128] = A@B + bias
// mode 1: atomicAdd column-sums of tanh(A@B + bias) into out[128]
// Block = 256 threads (8 waves); each wave owns a 16x16 tile (same 16 rows,
// wave-th group of 16 cols). Grid.x = N/16 (N % 16 == 0 -> EXEC all ones).
__global__ void k_gemm_wmma(const _Float16* __restrict__ A,
                            const _Float16* __restrict__ Bt,
                            const float* __restrict__ bias,
                            float* __restrict__ out,
                            int K, int mode) {
    const int wave = threadIdx.x >> 5;
    const int lane = threadIdx.x & 31;
    const int sub  = lane & 15;
    const int hi   = lane >> 4;
    const int row0 = blockIdx.x * 16;
    const int colb = wave * 16;

    const _Float16* arow = A  + (size_t)(row0 + sub) * K;
    const _Float16* brow = Bt + (size_t)(colb + sub) * K;

    v8f acc = {};
    for (int kb = 0; kb < K; kb += 32) {
        // A 16x32 f16 layout: lanes 0-15 -> K {kb..kb+7, kb+16..kb+23},
        //                     lanes16-31 -> K {kb+8..kb+15, kb+24..kb+31}
        v8h alo = *(const v8h*)(arow + kb + hi * 8);
        v8h ahi = *(const v8h*)(arow + kb + 16 + hi * 8);
        // B 32x16 f16 layout: lanes 0-15 -> K {kb..kb+7, kb+16..kb+23} (N=sub),
        //                     lanes16-31 -> K {kb+8..kb+15, kb+24..kb+31}
        v8h blo = *(const v8h*)(brow + kb + hi * 8);
        v8h bhi = *(const v8h*)(brow + kb + 16 + hi * 8);
        v16h a, b;
        #pragma unroll
        for (int i = 0; i < 8; ++i) {
            a[i] = alo[i]; a[8 + i] = ahi[i];
            b[i] = blo[i]; b[8 + i] = bhi[i];
        }
        acc = __builtin_amdgcn_wmma_f32_16x16x32_f16(
                  false, a, false, b, (short)0, acc, false, false);
    }

    const int col = colb + sub;
    if (mode == 0) {
        const float bv = bias[col];
        #pragma unroll
        for (int r = 0; r < 8; ++r) {
            int row = row0 + r + hi * 8;   // C/D layout: VGPR r -> M = r (+8 for hi lanes)
            out[(size_t)row * CD + col] = acc[r] + bv;
        }
    } else {
        const float bv = bias[col];
        float partial = 0.0f;
        #pragma unroll
        for (int r = 0; r < 8; ++r) partial += tanhf(acc[r] + bv);
        atomicAdd(&out[col], partial);
    }
}

// ---------------------------------------------------------------- attention coeffs
// a_src[et][n][h] = sum_d h[n,h*64+d]*att_src[et,h,d]  (same for dst)
__global__ void k_attn_coeff(const float* __restrict__ h,
                             const float* __restrict__ att_src,   // [2][2][64]
                             const float* __restrict__ att_dst,
                             float* __restrict__ asrc,            // [2][N][2]
                             float* __restrict__ adst) {
    int t = blockIdx.x * blockDim.x + threadIdx.x;
    if (t >= NN * HH) return;
    int n = t >> 1, hh = t & 1;
    const float* hp  = h + (size_t)n * CD + hh * DH;
    const float* s0p = att_src + hh * DH;
    const float* s1p = att_src + CD + hh * DH;   // et=1 offset = H*D = 128
    const float* d0p = att_dst + hh * DH;
    const float* d1p = att_dst + CD + hh * DH;
    float s0 = 0, s1 = 0, d0 = 0, d1 = 0;
    #pragma unroll 8
    for (int i = 0; i < DH; ++i) {
        float hv = hp[i];
        s0 += hv * s0p[i]; s1 += hv * s1p[i];
        d0 += hv * d0p[i]; d1 += hv * d1p[i];
    }
    asrc[t] = s0; asrc[(size_t)NN * HH + t] = s1;
    adst[t] = d0; adst[(size_t)NN * HH + t] = d1;
}

// ---------------------------------------------------------------- edge phase
__global__ void k_edge_max(const int* __restrict__ src, const int* __restrict__ dst,
                           const float* __restrict__ asrc, const float* __restrict__ adst,
                           float* __restrict__ amax, int E) {
    int e = blockIdx.x * blockDim.x + threadIdx.x;
    if (e >= E) return;
    int s = src[e], d = dst[e];
    #pragma unroll
    for (int h = 0; h < HH; ++h) {
        float a = asrc[(size_t)s * HH + h] + adst[(size_t)d * HH + h];
        a = (a > 0.0f) ? a : SLOPE * a;
        atomicMaxF(&amax[(size_t)d * HH + h], a);
    }
}

__global__ void k_edge_exp(const int* __restrict__ src, const int* __restrict__ dst,
                           const float* __restrict__ asrc, const float* __restrict__ adst,
                           const float* __restrict__ amax,
                           float* __restrict__ ebuf, float* __restrict__ ssum, int E) {
    int e = blockIdx.x * blockDim.x + threadIdx.x;
    if (e >= E) return;
    int s = src[e], d = dst[e];
    #pragma unroll
    for (int h = 0; h < HH; ++h) {
        float a = asrc[(size_t)s * HH + h] + adst[(size_t)d * HH + h];
        a = (a > 0.0f) ? a : SLOPE * a;
        float ex = expf(a - amax[(size_t)d * HH + h]);
        ebuf[(size_t)e * HH + h] = ex;
        atomicAdd(&ssum[(size_t)d * HH + h], ex);
    }
}

// one wave32 per edge; lane covers 4 channels (lanes 0-15 head 0, 16-31 head 1)
__global__ void k_edge_msg(const int* __restrict__ src, const int* __restrict__ dst,
                           const float* __restrict__ h,
                           const float* __restrict__ ebuf, const float* __restrict__ ssum,
                           float* __restrict__ out, int E) {
    int w    = (int)(((long)blockIdx.x * blockDim.x + threadIdx.x) >> 5);
    int lane = threadIdx.x & 31;
    if (w >= E) return;
    int s = src[w], d = dst[w];
    int hh = lane >> 4;
    float coeff = ebuf[(size_t)w * HH + hh] / (ssum[(size_t)d * HH + hh] + 1e-16f);
    float4 v = ((const float4*)(h + (size_t)s * CD))[lane];
    float* op = out + (size_t)d * CD + lane * 4;
    atomicAdd(op + 0, v.x * coeff);
    atomicAdd(op + 1, v.y * coeff);
    atomicAdd(op + 2, v.z * coeff);
    atomicAdd(op + 3, v.w * coeff);
}

// ---------------------------------------------------------------- semantic attn
__global__ void k_score(const float* __restrict__ colsum,  // [2][128] tanh col-sums
                        const float* __restrict__ q,
                        float* __restrict__ attn, float invN) {
    float s0 = 0.0f, s1 = 0.0f;
    for (int c = 0; c < CD; ++c) {
        s0 += q[c] * colsum[c] * invN;
        s1 += q[c] * colsum[CD + c] * invN;
    }
    float m  = fmaxf(s0, s1);
    float e0 = expf(s0 - m), e1 = expf(s1 - m);
    float inv = 1.0f / (e0 + e1);
    attn[0] = e0 * inv; attn[1] = e1 * inv;
}

__global__ void k_combine(const float* __restrict__ o0, const float* __restrict__ o1,
                          const float* __restrict__ attn, float* __restrict__ out, long n) {
    long i = (long)blockIdx.x * blockDim.x + threadIdx.x;
    if (i < n) out[i] = fmaxf(attn[0] * o0[i] + attn[1] * o1[i], 0.0f);
}

// ---------------------------------------------------------------- classifier
__global__ void k_linear(const float* __restrict__ h, const float* __restrict__ w,
                         const float* __restrict__ b, float* __restrict__ out, int N) {
    int n = blockIdx.x * blockDim.x + threadIdx.x;
    if (n >= N) return;
    float acc[OUTD];
    #pragma unroll
    for (int o = 0; o < OUTD; ++o) acc[o] = b[o];
    const float* hp = h + (size_t)n * CD;
    for (int c = 0; c < CD; ++c) {
        float hv = hp[c];
        #pragma unroll
        for (int o = 0; o < OUTD; ++o) acc[o] += hv * w[c * OUTD + o];
    }
    #pragma unroll
    for (int o = 0; o < OUTD; ++o) out[(size_t)n * OUTD + o] = acc[o];
}

// ---------------------------------------------------------------- launch
extern "C" void kernel_launch(void* const* d_in, const int* in_sizes, int n_in,
                              void* d_out, int out_size, void* d_ws, size_t ws_size,
                              hipStream_t stream) {
    const float* x      = (const float*)d_in[0];
    const int*   ei_sp  = (const int*)d_in[1];
    const int*   ei_si  = (const int*)d_in[2];
    const float* p_pw[2]  = { (const float*)d_in[3],  (const float*)d_in[10] };
    const float* p_pb[2]  = { (const float*)d_in[4],  (const float*)d_in[11] };
    const float* p_as[2]  = { (const float*)d_in[5],  (const float*)d_in[12] };
    const float* p_ad[2]  = { (const float*)d_in[6],  (const float*)d_in[13] };
    const float* p_q[2]   = { (const float*)d_in[7],  (const float*)d_in[14] };
    const float* p_kw[2]  = { (const float*)d_in[8],  (const float*)d_in[15] };
    const float* p_kb[2]  = { (const float*)d_in[9],  (const float*)d_in[16] };
    const float* lin_w  = (const float*)d_in[17];
    const float* lin_b  = (const float*)d_in[18];

    char* ws = (char*)d_ws;
    size_t off = 0;
    auto take = [&](size_t bytes) -> char* {
        char* p = ws + off;
        off += (bytes + 255) & ~(size_t)255;
        return p;
    };
    float*     bufA   = (float*)take((size_t)NN * CD * 4);   // layer in/out
    float*     bufH   = (float*)take((size_t)NN * CD * 4);   // projected h
    float*     bufO0  = (float*)take((size_t)NN * CD * 4);
    float*     bufO1  = (float*)take((size_t)NN * CD * 4);
    _Float16*  bufXh  = (_Float16*)take((size_t)NN * CD * 2);
    _Float16*  bufWt  = (_Float16*)take((size_t)CD * CD * 2);
    float*     asrc   = (float*)take((size_t)2 * NN * HH * 4);
    float*     adst   = (float*)take((size_t)2 * NN * HH * 4);
    float*     amax   = (float*)take((size_t)NN * HH * 4);
    float*     ssum   = (float*)take((size_t)NN * HH * 4);
    float*     ebuf   = (float*)take((size_t)EE * HH * 4);
    float*     colsum = (float*)take(2 * CD * 4);
    float*     attn   = (float*)take(2 * 4);

    const int*  eiarr[2] = { ei_sp, ei_si };
    float*      obuf[2]  = { bufO0, bufO1 };
    const dim3  B(256);

    auto run_layer = [&](const float* xin, int Kin, int L, float* outbuf) {
        // projection: h = xin @ W + b   (WMMA f16->f32)
        long nx = (long)NN * Kin;
        k_f32_to_f16<<<(unsigned)((nx + 255) / 256), B, 0, stream>>>(xin, bufXh, nx);
        k_wtrans_f16<<<(Kin * CD + 255) / 256, B, 0, stream>>>(p_pw[L], bufWt, Kin, CD);
        k_gemm_wmma<<<NN / 16, B, 0, stream>>>(bufXh, bufWt, p_pb[L], bufH, Kin, 0);
        k_attn_coeff<<<(NN * HH + 255) / 256, B, 0, stream>>>(bufH, p_as[L], p_ad[L], asrc, adst);

        for (int et = 0; et < 2; ++et) {
            const int* es = eiarr[et];
            const int* ed = eiarr[et] + EE;
            const float* aS = asrc + (size_t)et * NN * HH;
            const float* aD = adst + (size_t)et * NN * HH;
            k_fill<<<(NN * HH + 255) / 256, B, 0, stream>>>(amax, -INFINITY, NN * HH);
            (void)hipMemsetAsync(ssum, 0, (size_t)NN * HH * 4, stream);
            (void)hipMemsetAsync(obuf[et], 0, (size_t)NN * CD * 4, stream);
            k_edge_max<<<(EE + 255) / 256, B, 0, stream>>>(es, ed, aS, aD, amax, EE);
            k_edge_exp<<<(EE + 255) / 256, B, 0, stream>>>(es, ed, aS, aD, amax, ebuf, ssum, EE);
            k_edge_msg<<<(EE * 32) / 256, B, 0, stream>>>(es, ed, bufH, ebuf, ssum, obuf[et], EE);
            k_relu<<<(unsigned)(((long)NN * CD + 255) / 256), B, 0, stream>>>(obuf[et], (long)NN * CD);
        }

        // semantic attention: fused GEMM + tanh + column-sum
        (void)hipMemsetAsync(colsum, 0, 2 * CD * 4, stream);
        k_wtrans_f16<<<(CD * CD + 255) / 256, B, 0, stream>>>(p_kw[L], bufWt, CD, CD);
        for (int et = 0; et < 2; ++et) {
            long nn = (long)NN * CD;
            k_f32_to_f16<<<(unsigned)((nn + 255) / 256), B, 0, stream>>>(obuf[et], bufXh, nn);
            k_gemm_wmma<<<NN / 16, B, 0, stream>>>(bufXh, bufWt, p_kb[L], colsum + et * CD, CD, 1);
        }
        k_score<<<1, 1, 0, stream>>>(colsum, p_q[L], attn, 1.0f / (float)NN);
        k_combine<<<(unsigned)(((long)NN * CD + 255) / 256), B, 0, stream>>>(
            bufO0, bufO1, attn, outbuf, (long)NN * CD);
    };

    run_layer(x,    INF_, 0, bufA);
    run_layer(bufA, CD,   1, bufA);
    k_linear<<<(NN + 255) / 256, B, 0, stream>>>(bufA, lin_w, lin_b, (float*)d_out, NN);

    (void)in_sizes; (void)n_in; (void)out_size; (void)ws_size;
}
